// PointTransformerLayer_1881195676266
// MI455X (gfx1250) — matrix operand order repaired and checked by reading, not compiled, not run on hardware
//
#include <hip/hip_runtime.h>
#include <hip/hip_bf16.h>

#define N_PTS    50000
#define DIM      64
#define KNBR     16
#define QKV_LD   192
#define PPW      5     // points per wave; 1250 blocks * 8 waves * 5 = 50000 exact

typedef __attribute__((ext_vector_type(16))) __bf16 v16bf;
typedef __attribute__((ext_vector_type(8)))  float  v8f;

// ---------------------------------------------------------------------------
// WMMA wrapper (CDNA5 wave32, V_WMMA_F32_16X16X32_BF16)
// ---------------------------------------------------------------------------
__device__ __forceinline__ v8f wmma_bf16(v16bf a, v16bf b, v8f c) {
  return __builtin_amdgcn_wmma_f32_16x16x32_bf16(false, a, false, b, (short)0, c,
                                                 false, false);
}

// Operand tile = 512 bf16; lane's 16 elements contiguous.  lanePtr is
// precomputed (buffer + lane*16); tile index compile-time -> 2x ds_load_b128.
__device__ __forceinline__ v16bf load_tileL(const __bf16* lanePtr, int tile) {
  return *(const v16bf*)(lanePtr + tile * 512);
}

// Convert fp32 row-major W[Kdim][Ndim] into B-operand tiles ([n-tile][k-chunk]).
__device__ __forceinline__ void pack_B(const float* __restrict__ W, __bf16* pk,
                                       int Kdim, int Ndim, int tid, int nthreads) {
  const int KC = Kdim >> 5;
  const int total = Kdim * Ndim;
  for (int i = tid; i < total; i += nthreads) {
    int tile = i >> 9;
    int within = i & 511;
    int lanep = within >> 4;
    int e = within & 15;
    int kc = tile % KC;
    int nt = tile / KC;
    int k = kc * 32 + (lanep >> 4) * 8 + (e & 7) + (e >> 3) * 16;
    int n = nt * 16 + (lanep & 15);
    pk[i] = (__bf16)W[k * Ndim + n];
  }
}

// lane xor-1 exchange via DPP quad_perm(1,0,3,2) -- pure VALU, no LDS.
__device__ __forceinline__ float dpp_xor1_f(float v) {
  int r = __builtin_amdgcn_update_dpp(0, __builtin_bit_cast(int, v),
                                      0xB1, 0xF, 0xF, true);
  return __builtin_bit_cast(float, r);
}

__device__ __forceinline__ unsigned int pack2bf(float lo, float hi) {
  unsigned short a = __builtin_bit_cast(unsigned short, (__bf16)lo);
  unsigned short b = __builtin_bit_cast(unsigned short, (__bf16)hi);
  return ((unsigned int)b << 16) | (unsigned int)a;
}

// Per-lane-invariant element offset into an A-packed staging buffer.
// Full A-layout address = laneOff + (nt>>1)*512 + r*16 + 8*(nt&1)  (elements).
__device__ __forceinline__ int apk_lane_off(int lane) {
  int l15 = lane & 15, lhalf = lane >> 4;
  int halfp = (l15 >> 3) & 1;
  return (8 * lhalf + 16 * halfp) * 16 + (l15 & 6);
}

// Store one C-layout value (row m=r+8*(lane>>4), col f=(lane&15)+16*nt) into
// the A-packed buffer; column pairs exchanged via DPP, both lanes of a pair
// write the identical b32 to the identical LDS address (immediate offsets).
__device__ __forceinline__ void storeA_pair(__bf16* lanePtr, int lane, int r,
                                            int nt, float val) {
  float other = dpp_xor1_f(val);
  float lo = (lane & 1) ? other : val;
  float hi = (lane & 1) ? val : other;
  *(unsigned int*)(lanePtr + ((nt >> 1) * 512 + r * 16 + 8 * (nt & 1))) =
      pack2bf(lo, hi);
}

// ---------------------------------------------------------------------------
// Kernel 0: qkv = x @ w_qkv ([N,64]@[64,192] -> fp32 qkv in d_ws).
// N % 16 == 0 -> every tile full, no guards.
// ---------------------------------------------------------------------------
__global__ void qkv_kernel(const float* __restrict__ x,
                           const float* __restrict__ w_qkv,
                           float* __restrict__ qkv) {
  __shared__ __bf16 wpk[DIM * QKV_LD];  // 24 KB, packed B tiles [12 nt][2 kc]
  pack_B(w_qkv, wpk, DIM, QKV_LD, threadIdx.x, blockDim.x);
  __syncthreads();

  int wave = threadIdx.x >> 5, lane = threadIdx.x & 31;
  int row0 = (blockIdx.x * 8 + wave) * 16;
  if (row0 >= N_PTS) return;

  int m = lane & 15, half = lane >> 4;
  const __bf16* wl = wpk + lane * 16;

  v16bf a0, a1;
  const float* p = x + (row0 + m) * DIM + 8 * half;
#pragma unroll
  for (int j = 0; j < 8; ++j) {
    a0[j]     = (__bf16)p[j];
    a0[8 + j] = (__bf16)p[16 + j];
    a1[j]     = (__bf16)p[32 + j];
    a1[8 + j] = (__bf16)p[48 + j];
  }

  float* ob = qkv + (row0 + 8 * half) * QKV_LD + m;
#pragma unroll
  for (int nt = 0; nt < 12; ++nt) {
    v8f c = {};
    c = wmma_bf16(a0, load_tileL(wl, nt * 2 + 0), c);
    c = wmma_bf16(a1, load_tileL(wl, nt * 2 + 1), c);
#pragma unroll
    for (int rr = 0; rr < 8; ++rr)
      ob[rr * QKV_LD + nt * 16] = c[rr];
  }
}

// ---------------------------------------------------------------------------
// Kernel 0b: one-shot pack of MLP weights into bf16 operand tiles in d_ws,
// so ptl blocks can bulk-copy ready bytes instead of converting per block.
// ---------------------------------------------------------------------------
#define SMEM_W_AMW1   (64 * 256)
#define SMEM_W_AMW2   (256 * 64)
#define SMEM_W_PMW2   (64 * 64)
#define W_ELEMS       (SMEM_W_AMW1 + SMEM_W_AMW2 + SMEM_W_PMW2)   // 36864
#define SMEM_WEIGHT_BYTES (W_ELEMS * 2)                           // 73728
#define SMEM_PERWAVE_BYTES ((16 * 64 + 16 * 256) * 2)             // 10240
#define SMEM_TOTAL (SMEM_WEIGHT_BYTES + 8 * SMEM_PERWAVE_BYTES)   // 155648

__global__ void pack_weights_kernel(const float* __restrict__ am_w1,
                                    const float* __restrict__ am_w2,
                                    const float* __restrict__ pm_w2,
                                    __bf16* __restrict__ gw) {
  int tid = blockIdx.x * blockDim.x + threadIdx.x;
  int nth = gridDim.x * blockDim.x;
  pack_B(am_w1, gw, 64, 256, tid, nth);
  pack_B(am_w2, gw + SMEM_W_AMW1, 256, 64, tid, nth);
  pack_B(pm_w2, gw + SMEM_W_AMW1 + SMEM_W_AMW2, 64, 64, tid, nth);
}

// ---------------------------------------------------------------------------
// Kernel 1: fused per-point layer.  One wave handles PPW consecutive points.
// Weight tiles arrive via GLOBAL_LOAD_ASYNC_TO_LDS_B128 (ASYNCcnt-tracked).
// ---------------------------------------------------------------------------
typedef __attribute__((address_space(3))) char* lds_ptr_t;

__global__ void ptl_kernel(const float* __restrict__ pos,
                           const int* __restrict__ attn_index,
                           const unsigned char* __restrict__ mask,
                           const float* __restrict__ pm_w1,
                           const float* __restrict__ pm_b1,
                           const float* __restrict__ pm_b2,
                           const float* __restrict__ am_b1,
                           const float* __restrict__ am_b2,
                           const __bf16* __restrict__ gw,   // packed weight tiles
                           const float* __restrict__ qkv,
                           float* __restrict__ out) {
  extern __shared__ char smem[];
  __bf16* s_am_w1 = (__bf16*)smem;                 // 32 tiles: [16 nt][2 kc]
  __bf16* s_am_w2 = s_am_w1 + SMEM_W_AMW1;         // 32 tiles: [4 nt][8 kc]
  __bf16* s_pm_w2 = s_am_w2 + SMEM_W_AMW2;         // 8 tiles:  [4 nt][2 kc]
  char* wavebase  = (char*)(s_pm_w2 + SMEM_W_PMW2);

  int t = threadIdx.x, wave = t >> 5, lane = t & 31;

  // ---- async bulk copy of packed weights: 18 iters * 256 lanes * 16 B = 73728 B
  {
    unsigned ldsbase = (unsigned)(unsigned long long)(lds_ptr_t)smem;
    unsigned dst = ldsbase + (unsigned)(t * 16);
    unsigned long long src = (unsigned long long)((const char*)gw + t * 16);
    for (int i = 0; i < 18; ++i) {
      asm volatile("global_load_async_to_lds_b128 %0, %1, off"
                   :: "v"(dst), "v"(src) : "memory");
      dst += 4096;
      src += 4096;
    }
    asm volatile("s_wait_asynccnt 0x0" ::: "memory");
  }
  __syncthreads();

  __bf16* actA = (__bf16*)(wavebase + wave * SMEM_PERWAVE_BYTES);  // 2 chunks
  __bf16* actB = actA + 16 * 64;                                   // 8 chunks

  int m = lane & 15, half = lane >> 4;

  // Precomputed lane pointers (all subsequent LDS ops use immediate offsets).
  int lo16 = lane * 16;
  int loA  = apk_lane_off(lane);
  const __bf16* w1l = s_am_w1 + lo16;
  const __bf16* w2l = s_am_w2 + lo16;
  const __bf16* pwl = s_pm_w2 + lo16;
  const __bf16* aAl = actA + lo16;
  const __bf16* aBl = actB + lo16;
  __bf16* sAl = actA + loA;
  __bf16* sBl = actB + loA;

  // Per-lane weight taps / biases (loop-invariant across the PPW points).
  float w1v[3][4], b1v[4], b2v[4], ab2v[4], ab1v[16];
#pragma unroll
  for (int tt = 0; tt < 4; ++tt) {
    w1v[0][tt] = pm_w1[0 * 64 + m + 16 * tt];
    w1v[1][tt] = pm_w1[1 * 64 + m + 16 * tt];
    w1v[2][tt] = pm_w1[2 * 64 + m + 16 * tt];
    b1v[tt]    = pm_b1[m + 16 * tt];
    b2v[tt]    = pm_b2[m + 16 * tt];
    ab2v[tt]   = am_b2[m + 16 * tt];
  }
#pragma unroll
  for (int nt = 0; nt < 16; ++nt) ab1v[nt] = am_b1[m + 16 * nt];

  const float NEG_INF = -__builtin_inff();
  int base_n = (blockIdx.x * 8 + wave) * PPW;  // exact tiling, no guards

#pragma unroll 1
  for (int it = 0; it < PPW; ++it) {
    int n = base_n + it;

    // Neighbor indices + gather bases; prefetch k/v rows (L2-resident).
    int idxr[8];
    const float* krow[8];
    const float* vrow[8];
#pragma unroll
    for (int r = 0; r < 8; ++r) {
      idxr[r] = attn_index[n * KNBR + r + 8 * half];
      const float* rp = qkv + idxr[r] * QKV_LD + m;
      krow[r] = rp + 64;
      vrow[r] = rp + 128;
      __builtin_prefetch(krow[r], 0, 3);
      __builtin_prefetch(vrow[r], 0, 3);
    }
    const float* qrow = qkv + n * QKV_LD + m;

    float cx = pos[n * 3 + 0], cy = pos[n * 3 + 1], cz = pos[n * 3 + 2];

    // ---- pos-MLP layer 1 (VALU): h1 = relu(rel_pos @ pm_w1 + b1) -> actA
#pragma unroll
    for (int r = 0; r < 8; ++r) {
      float px = pos[idxr[r] * 3 + 0] - cx;
      float py = pos[idxr[r] * 3 + 1] - cy;
      float pz = pos[idxr[r] * 3 + 2] - cz;
#pragma unroll
      for (int tt = 0; tt < 4; ++tt) {
        float h = px * w1v[0][tt] + py * w1v[1][tt] + pz * w1v[2][tt] + b1v[tt];
        storeA_pair(sAl, lane, r, tt, fmaxf(h, 0.f));
      }
    }
    asm volatile("s_wait_dscnt 0x0" ::: "memory");

    // ---- rel = h1 @ pm_w2 + b2  (kept in C-layout registers)
    float relv[4][8];
    {
      v16bf a0 = load_tileL(aAl, 0);
      v16bf a1 = load_tileL(aAl, 1);
#pragma unroll
      for (int nt = 0; nt < 4; ++nt) {
        v8f c = {};
        c = wmma_bf16(a0, load_tileL(pwl, nt * 2 + 0), c);
        c = wmma_bf16(a1, load_tileL(pwl, nt * 2 + 1), c);
#pragma unroll
        for (int r = 0; r < 8; ++r) relv[nt][r] = c[r] + b2v[nt];
      }
    }

    // ---- w_in = x_k - x_q + rel -> actA (immediate-offset gathers)
#pragma unroll
    for (int nt = 0; nt < 4; ++nt) {
      float xq = qrow[16 * nt];
#pragma unroll
      for (int r = 0; r < 8; ++r) {
        float xk = krow[r][16 * nt];
        storeA_pair(sAl, lane, r, nt, xk - xq + relv[nt][r]);
      }
    }
    asm volatile("s_wait_dscnt 0x0" ::: "memory");

    // ---- h2 = relu(w_in @ am_w1 + b1) -> actB (A-packed, 8 chunks)
    {
      v16bf wa0 = load_tileL(aAl, 0);
      v16bf wa1 = load_tileL(aAl, 1);
#pragma unroll
      for (int nt = 0; nt < 16; ++nt) {
        v8f c = {};
        c = wmma_bf16(wa0, load_tileL(w1l, nt * 2 + 0), c);
        c = wmma_bf16(wa1, load_tileL(w1l, nt * 2 + 1), c);
#pragma unroll
        for (int r = 0; r < 8; ++r)
          storeA_pair(sBl, lane, r, nt, fmaxf(c[r] + ab1v[nt], 0.f));
      }
    }
    asm volatile("s_wait_dscnt 0x0" ::: "memory");

    // ---- attn = h2 @ am_w2 + b2  (4 N-tiles x 8 K-chunks)
    v8f attn[4];
    {
      v16bf ha[8];
#pragma unroll
      for (int i = 0; i < 8; ++i) ha[i] = load_tileL(aBl, i);
#pragma unroll
      for (int nt = 0; nt < 4; ++nt) {
        v8f c = {};
#pragma unroll
        for (int i = 0; i < 8; ++i)
          c = wmma_bf16(ha[i], load_tileL(w2l, nt * 8 + i), c);
#pragma unroll
        for (int r = 0; r < 8; ++r) c[r] += ab2v[nt];
        attn[nt] = c;
      }
    }

    // ---- masked softmax over 16 neighbors (8 rows local + 8 in lane^16)
    unsigned char mk[8];
#pragma unroll
    for (int r = 0; r < 8; ++r) mk[r] = mask[n * KNBR + r + 8 * half];
#pragma unroll
    for (int nt = 0; nt < 4; ++nt) {
#pragma unroll
      for (int r = 0; r < 8; ++r)
        if (mk[r]) attn[nt][r] = NEG_INF;
      float mx = attn[nt][0];
#pragma unroll
      for (int r = 1; r < 8; ++r) mx = fmaxf(mx, attn[nt][r]);
      mx = fmaxf(mx, __shfl_xor(mx, 16, 32));
      float s = 0.f;
#pragma unroll
      for (int r = 0; r < 8; ++r) {
        float e = __expf(attn[nt][r] - mx);
        attn[nt][r] = e;
        s += e;
      }
      s += __shfl_xor(s, 16, 32);
      float inv = 1.f / s;
#pragma unroll
      for (int r = 0; r < 8; ++r) attn[nt][r] *= inv;
    }

    // ---- agg = sum_k (x_v + rel) * w -> out[n, :]
    float accv[4];
#pragma unroll
    for (int nt = 0; nt < 4; ++nt) {
      float acc = 0.f;
#pragma unroll
      for (int r = 0; r < 8; ++r) {
        float xv = vrow[r][16 * nt];
        acc += (xv + relv[nt][r]) * attn[nt][r];
      }
      acc += __shfl_xor(acc, 16, 32);
      accv[nt] = acc;
    }
    if (half == 0) {
      float* ob = out + n * DIM + m;
      ob[0]  = accv[0];
      ob[16] = accv[1];
      ob[32] = accv[2];
      ob[48] = accv[3];
    }
  }
}

// ---------------------------------------------------------------------------
// Host launcher
// ---------------------------------------------------------------------------
extern "C" void kernel_launch(void* const* d_in, const int* in_sizes, int n_in,
                              void* d_out, int out_size, void* d_ws, size_t ws_size,
                              hipStream_t stream) {
  const float* x          = (const float*)d_in[0];
  const float* pos        = (const float*)d_in[1];
  const int*   attn_index = (const int*)d_in[2];
  const unsigned char* mask = (const unsigned char*)d_in[3];
  const float* w_qkv      = (const float*)d_in[4];
  const float* pm_w1      = (const float*)d_in[5];
  const float* pm_b1      = (const float*)d_in[6];
  const float* pm_w2      = (const float*)d_in[7];
  const float* pm_b2      = (const float*)d_in[8];
  const float* am_w1      = (const float*)d_in[9];
  const float* am_b1      = (const float*)d_in[10];
  const float* am_w2      = (const float*)d_in[11];
  const float* am_b2      = (const float*)d_in[12];
  float* out = (float*)d_out;

  float*  qkv = (float*)d_ws;                           // 38.4 MB
  __bf16* gw  = (__bf16*)((char*)d_ws + (size_t)N_PTS * QKV_LD * 4);  // 73.7 KB

  (void)hipFuncSetAttribute((const void*)ptl_kernel,
                            hipFuncAttributeMaxDynamicSharedMemorySize,
                            SMEM_TOTAL);

  int blocks0 = (N_PTS / 16 + 7) / 8;   // 391 blocks, 16 rows per wave
  qkv_kernel<<<blocks0, 256, 0, stream>>>(x, w_qkv, qkv);

  pack_weights_kernel<<<16, 256, 0, stream>>>(am_w1, am_w2, pm_w2, gw);

  int blocks1 = N_PTS / (8 * PPW);      // 1250 blocks, exact
  ptl_kernel<<<blocks1, 256, SMEM_TOTAL, stream>>>(
      pos, attn_index, mask, pm_w1, pm_b1, pm_b2, am_b1, am_b2, gw, qkv, out);
}
